// ContextualLoss_74371653697716
// MI455X (gfx1250) — compile-verified, matrix operand order checked by reading, not run on hardware
//
#include <hip/hip_runtime.h>
#include <hip/hip_bf16.h>

// ---- types for WMMA / async ----
typedef __attribute__((ext_vector_type(16))) _Float16 v16h;
typedef __attribute__((ext_vector_type(8)))  _Float16 v8h;
typedef __attribute__((ext_vector_type(8)))  float    v8f;
typedef __attribute__((ext_vector_type(4)))  int      v4i;

#define BATCH 4
#define CH    256
#define HW    4096
#define NT    (HW / 16)          // 256 16-wide tiles per sweep
#define LOG2E 1.4426950408889634f

#if __has_builtin(__builtin_amdgcn_global_load_async_to_lds_b128) && \
    __has_builtin(__builtin_amdgcn_s_wait_asynccnt)
#define HAVE_ASYNC_LDS 1
typedef __attribute__((address_space(1))) v4i gv4i;   // global b128 chunk
typedef __attribute__((address_space(3))) v4i lv4i;   // LDS b128 chunk
#else
#define HAVE_ASYNC_LDS 0
#endif

// Load one 16x32 f16 WMMA fragment for this lane from a row-major [*,256] row.
// ISA layout (16-bit A 16x32): lanes 0-15 hold K = f*32 + {0..7} and f*32+16+{0..7},
// lanes 16-31 the same shifted by 8.  B (32x16) is symmetric with N = lane%16.
__device__ __forceinline__ v16h load_frag(const _Float16* rowbase, int f, int kg) {
    v8h lo = *(const v8h*)(rowbase + f * 32 + kg);
    v8h hi = *(const v8h*)(rowbase + f * 32 + 16 + kg);
    return __builtin_shufflevector(lo, hi, 0,1,2,3,4,5,6,7,8,9,10,11,12,13,14,15);
}

// Stage a 16x256-half tile (8 KB) into LDS: 256 threads x 2 chunks of 16 B.
// Async path: GLOBAL_LOAD_ASYNC_TO_LDS_B128 (ASYNCcnt), no VGPR round-trip.
__device__ __forceinline__ void stage_tile(_Float16* dstLds, const _Float16* src,
                                           int tid) {
#pragma unroll
    for (int ch = 0; ch < 2; ++ch) {
        int c2 = tid + ch * 256;
        int off = ((c2 >> 5) * CH) + ((c2 & 31) << 3);
#if HAVE_ASYNC_LDS
        __builtin_amdgcn_global_load_async_to_lds_b128(
            (gv4i*)(src + off), (lv4i*)(dstLds + off), 0, 0);
#else
        *(v8h*)(dstLds + off) = *(const v8h*)(src + off);
#endif
    }
}

// Wait for all but the most recent stage_tile (2 async ops/wave) to land.
__device__ __forceinline__ void wait_stage_minus_one() {
#if HAVE_ASYNC_LDS
    __builtin_amdgcn_s_wait_asynccnt(2);
#endif
}
__device__ __forceinline__ void wait_stage_all() {
#if HAVE_ASYNC_LDS
    __builtin_amdgcn_s_wait_asynccnt(0);
#endif
}

// ---------------- kernel 1: per-channel mean of targets over (B,H,W) -------------
__global__ __launch_bounds__(256) void cx_chanmean(const float* __restrict__ tgt,
                                                   float* __restrict__ meanT) {
    __shared__ float red[256];
    const int c = blockIdx.x, tid = threadIdx.x;
    float s = 0.f;
    for (int n = tid; n < BATCH * HW; n += 256) {
        int b = n >> 12, sp = n & (HW - 1);
        s += tgt[((size_t)(b * CH + c)) * HW + sp];
    }
    red[tid] = s; __syncthreads();
    for (int off = 128; off > 0; off >>= 1) {
        if (tid < off) red[tid] += red[tid + off];
        __syncthreads();
    }
    if (tid == 0) meanT[c] = red[0] * (1.0f / (BATCH * HW));
}

// ------------- kernel 2: center, l2-normalize over C, pack f16 [B,HW,C] ----------
__global__ __launch_bounds__(256) void cx_normalize(const float* __restrict__ inp,
                                                    const float* __restrict__ tgt,
                                                    const float* __restrict__ meanT,
                                                    _Float16* __restrict__ Xn,
                                                    _Float16* __restrict__ Tn) {
    __shared__ float rx[256], rt[256];
    const int tid = threadIdx.x;                 // tid == channel
    const int b = blockIdx.x >> 12;
    const int s = blockIdx.x & (HW - 1);
    const size_t idx = ((size_t)(b * CH + tid)) * HW + s;
    const float mu = meanT[tid];
    const float x = inp[idx] - mu;
    const float t = tgt[idx] - mu;
    rx[tid] = x * x; rt[tid] = t * t;
    __syncthreads();
    for (int off = 128; off > 0; off >>= 1) {
        if (tid < off) { rx[tid] += rx[tid + off]; rt[tid] += rt[tid + off]; }
        __syncthreads();
    }
    const float rnx = __builtin_amdgcn_rsqf(rx[0]);
    const float rnt = __builtin_amdgcn_rsqf(rt[0]);
    const size_t o = ((size_t)b * HW + s) * CH + tid;
    Xn[o] = (_Float16)(x * rnx);
    Tn[o] = (_Float16)(t * rnt);
}

// --- kernel 3: per-row stats. Each wave owns a 16-row i-strip (A in registers),
// --- sweeps all j-tiles twice: (1) row-min of cd, (2) row-sum of exp weights.
// --- Double-buffered LDS tile + async global->LDS staging pipelined vs WMMA. ---
__global__ __launch_bounds__(256) void cx_rowstats(const _Float16* __restrict__ Xn,
                                                   const _Float16* __restrict__ Tn,
                                                   float* __restrict__ dminG,
                                                   float* __restrict__ rsumG) {
    __shared__ __align__(16) _Float16 tB[2][16 * CH];   // 2 x 8 KB j-tiles
    const int tid   = threadIdx.x;
    const int b     = blockIdx.x >> 5;               // 32 blocks per batch
    const int iblk  = blockIdx.x & 31;
    const int wave  = tid >> 5;
    const int lane  = tid & 31;
    const int istrip = iblk * 128 + wave * 16;
    const int mn    = lane & 15;
    const int kg    = (lane >> 4) << 3;
    const size_t bbase = (size_t)b * HW;
    const _Float16* Tb = Tn + bbase * CH;

    // A strip (16 rows x K=256) held in registers: 8 fragments of K=32
    v16h afrag[8];
    {
        const _Float16* arow = Xn + (bbase + istrip + mn) * CH;
        #pragma unroll
        for (int f = 0; f < 8; ++f) afrag[f] = load_frag(arow, f, kg);
    }

    float rmin[8];
    #pragma unroll
    for (int r = 0; r < 8; ++r) rmin[r] = 3.0e38f;

    // ---- sweep 1: row-min of cosine distance ----
    stage_tile(tB[0], Tb, tid);
    for (int jt = 0; jt < NT; ++jt) {
        const int cur = jt & 1;
        if (jt + 1 < NT) {
            stage_tile(tB[cur ^ 1], Tb + (size_t)(jt + 1) * 16 * CH, tid);
            wait_stage_minus_one();
        } else {
            wait_stage_all();
        }
        __syncthreads();
        v8f acc = {0.f,0.f,0.f,0.f,0.f,0.f,0.f,0.f};
        const _Float16* brow = tB[cur] + mn * CH;
        #pragma unroll
        for (int f = 0; f < 8; ++f) {
            v16h bf = load_frag(brow, f, kg);
            acc = __builtin_amdgcn_wmma_f32_16x16x32_f16(false, afrag[f], false, bf,
                                                         (short)0, acc, false, false);
        }
        #pragma unroll
        for (int r = 0; r < 8; ++r)
            rmin[r] = fminf(rmin[r], (1.0f - acc[r]) * 0.5f);
        __syncthreads();
    }

    // reduce min across the 16-lane group; every lane ends with its row's min
    float t10[8], dmv[8];
    #pragma unroll
    for (int r = 0; r < 8; ++r) {
        float m = rmin[r];
        m = fminf(m, __shfl_xor(m, 1));
        m = fminf(m, __shfl_xor(m, 2));
        m = fminf(m, __shfl_xor(m, 4));
        m = fminf(m, __shfl_xor(m, 8));
        dmv[r] = m;
        t10[r] = 10.0f / (m + 1e-5f);      // (1/h) / (min + eps)
    }

    float wsum[8];
    #pragma unroll
    for (int r = 0; r < 8; ++r) wsum[r] = 0.f;

    // ---- sweep 2: row-sum of w = exp((1 - rel)/h) ----
    stage_tile(tB[0], Tb, tid);
    for (int jt = 0; jt < NT; ++jt) {
        const int cur = jt & 1;
        if (jt + 1 < NT) {
            stage_tile(tB[cur ^ 1], Tb + (size_t)(jt + 1) * 16 * CH, tid);
            wait_stage_minus_one();
        } else {
            wait_stage_all();
        }
        __syncthreads();
        v8f acc = {0.f,0.f,0.f,0.f,0.f,0.f,0.f,0.f};
        const _Float16* brow = tB[cur] + mn * CH;
        #pragma unroll
        for (int f = 0; f < 8; ++f) {
            v16h bf = load_frag(brow, f, kg);
            acc = __builtin_amdgcn_wmma_f32_16x16x32_f16(false, afrag[f], false, bf,
                                                         (short)0, acc, false, false);
        }
        #pragma unroll
        for (int r = 0; r < 8; ++r) {
            float cd = (1.0f - acc[r]) * 0.5f;
            wsum[r] += __builtin_amdgcn_exp2f((10.0f - cd * t10[r]) * LOG2E);
        }
        __syncthreads();
    }

    #pragma unroll
    for (int r = 0; r < 8; ++r) {
        float s = wsum[r];
        s += __shfl_xor(s, 1);
        s += __shfl_xor(s, 2);
        s += __shfl_xor(s, 4);
        s += __shfl_xor(s, 8);
        if ((lane & 15) == 0) {
            int gi = istrip + r + ((lane >> 4) << 3);
            dminG[bbase + gi] = dmv[r];
            rsumG[bbase + gi] = s;
        }
    }
}

// --- kernel 4: per-column max of cs = w[i,j]/rowsum[i]. Wave owns a 16-col j
// --- strip (B in registers), sweeps all i-tiles staged async through LDS. ------
__global__ __launch_bounds__(256) void cx_colmax(const _Float16* __restrict__ Xn,
                                                 const _Float16* __restrict__ Tn,
                                                 const float* __restrict__ dminG,
                                                 const float* __restrict__ rsumG,
                                                 float* __restrict__ colmaxG) {
    __shared__ __align__(16) _Float16 tA[2][16 * CH];
    __shared__ float sD[2][16], sR[2][16];
    const int tid   = threadIdx.x;
    const int b     = blockIdx.x >> 5;
    const int jblk  = blockIdx.x & 31;
    const int wave  = tid >> 5;
    const int lane  = tid & 31;
    const int jstrip = jblk * 128 + wave * 16;
    const int mn    = lane & 15;
    const int kg    = (lane >> 4) << 3;
    const size_t bbase = (size_t)b * HW;
    const _Float16* Xb = Xn + bbase * CH;

    // B strip (K=256 x 16 cols) in registers; lane holds column N = lane%16
    v16h bfrag[8];
    {
        const _Float16* brow = Tn + (bbase + jstrip + mn) * CH;
        #pragma unroll
        for (int f = 0; f < 8; ++f) bfrag[f] = load_frag(brow, f, kg);
    }

    float cmax = 0.f;
    stage_tile(tA[0], Xb, tid);
    if (tid < 16) {
        sD[0][tid] = dminG[bbase + tid];
        sR[0][tid] = rsumG[bbase + tid];
    }
    for (int it = 0; it < NT; ++it) {
        const int cur = it & 1;
        if (it + 1 < NT) {
            stage_tile(tA[cur ^ 1], Xb + (size_t)(it + 1) * 16 * CH, tid);
            if (tid < 16) {
                sD[cur ^ 1][tid] = dminG[bbase + (it + 1) * 16 + tid];
                sR[cur ^ 1][tid] = rsumG[bbase + (it + 1) * 16 + tid];
            }
            wait_stage_minus_one();
        } else {
            wait_stage_all();
        }
        __syncthreads();
        v8f acc = {0.f,0.f,0.f,0.f,0.f,0.f,0.f,0.f};
        const _Float16* arow = tA[cur] + mn * CH;
        #pragma unroll
        for (int f = 0; f < 8; ++f) {
            v16h af = load_frag(arow, f, kg);
            acc = __builtin_amdgcn_wmma_f32_16x16x32_f16(false, af, false, bfrag[f],
                                                         (short)0, acc, false, false);
        }
        #pragma unroll
        for (int r = 0; r < 8; ++r) {
            int m = r + ((lane >> 4) << 3);          // local i row in this tile
            float cd = (1.0f - acc[r]) * 0.5f;
            float e  = (10.0f - cd * (10.0f / (sD[cur][m] + 1e-5f))) * LOG2E;
            float cs = __builtin_amdgcn_exp2f(e) / (sR[cur][m] + 1e-8f);
            cmax = fmaxf(cmax, cs);
        }
        __syncthreads();
    }
    // combine the two half-wave M groups, then lanes 0..15 write one column each
    cmax = fmaxf(cmax, __shfl_xor(cmax, 16));
    if (lane < 16) colmaxG[bbase + jstrip + lane] = cmax;
}

// ---------------- kernel 5: CS = mean_j colmax, loss = mean_b(-log CS) -----------
__global__ __launch_bounds__(256) void cx_final(const float* __restrict__ colmaxG,
                                                float* __restrict__ out) {
    __shared__ float red[256];
    const int tid = threadIdx.x;
    float loss = 0.f;
    for (int b = 0; b < BATCH; ++b) {
        float s = 0.f;
        for (int j = tid; j < HW; j += 256) s += colmaxG[b * HW + j];
        red[tid] = s; __syncthreads();
        for (int off = 128; off > 0; off >>= 1) {
            if (tid < off) red[tid] += red[tid + off];
            __syncthreads();
        }
        float CS = red[0] * (1.0f / HW);
        loss += -0.69314718055994531f * __builtin_amdgcn_logf(CS);  // -ln = -log2*ln2
        __syncthreads();
    }
    if (tid == 0) out[0] = loss * (1.0f / BATCH);
}

extern "C" void kernel_launch(void* const* d_in, const int* in_sizes, int n_in,
                              void* d_out, int out_size, void* d_ws, size_t ws_size,
                              hipStream_t stream) {
    const float* inp = (const float*)d_in[0];
    const float* tgt = (const float*)d_in[1];
    char* ws = (char*)d_ws;

    // workspace layout (~16.3 MB total)
    float*    meanT   = (float*)(ws + 0);                       //  1 KB
    float*    dminG   = (float*)(ws + 1024);                    // 64 KB
    float*    rsumG   = (float*)(ws + 1024 + 65536);            // 64 KB
    float*    colmaxG = (float*)(ws + 1024 + 2 * 65536);        // 64 KB
    _Float16* Xn      = (_Float16*)(ws + 262144);               //  8 MB
    _Float16* Tn      = (_Float16*)(ws + 262144 + 8388608);     //  8 MB
    float*    out     = (float*)d_out;

    cx_chanmean <<<CH,          256, 0, stream>>>(tgt, meanT);
    cx_normalize<<<BATCH * HW,  256, 0, stream>>>(inp, tgt, meanT, Xn, Tn);
    cx_rowstats <<<BATCH * 32,  256, 0, stream>>>(Xn, Tn, dminG, rsumG);
    cx_colmax   <<<BATCH * 32,  256, 0, stream>>>(Xn, Tn, dminG, rsumG, colmaxG);
    cx_final    <<<1,           256, 0, stream>>>(colmaxG, out);
}
